// CrossAttentionBlock_80453327389402
// MI455X (gfx1250) — matrix operand order repaired and checked by reading, not compiled, not run on hardware
//
#include <hip/hip_runtime.h>
#include <hip/hip_bf16.h>

typedef __attribute__((ext_vector_type(16))) _Float16 v16h;
typedef __attribute__((ext_vector_type(8)))  float    v8f;

#define D_MODEL 128   // ATT_DIM

// ---------------------------------------------------------------------------
// WMMA helper: D = A(16x32 f16) * B(32x16 f16) + C(16x16 f32)
// ---------------------------------------------------------------------------
__device__ __forceinline__ v8f wmma_f16(v16h a, v16h b, v8f c) {
  return __builtin_amdgcn_wmma_f32_16x16x32_f16(false, a, false, b,
                                                (short)0, c, false, false);
}

// ---------------------------------------------------------------------------
// Fragment loaders following ISA 7.12.2 lane layouts (wave32).
// A (16x32, MxK): lane holds row M = lane%16; element i holds
//   K = (i<8) ? 8*hf + i : 16 + 8*hf + (i-8), where hf = lane/16.
// B (32x16, KxN): lane holds col N = lane%16; element j holds same K as A.
// ---------------------------------------------------------------------------
__device__ __forceinline__ v16h load_a_f32(const float* __restrict__ row, int kbase, int hf) {
  v16h a;
  const float* p0 = row + kbase + 8 * hf;
  const float* p1 = row + kbase + 16 + 8 * hf;
#pragma unroll
  for (int i = 0; i < 8; ++i) { a[i] = (_Float16)p0[i]; a[8 + i] = (_Float16)p1[i]; }
  return a;
}

__device__ __forceinline__ v16h load_a_f16(const _Float16* __restrict__ row, int kbase, int hf) {
  v16h a;
  const _Float16* p0 = row + kbase + 8 * hf;
  const _Float16* p1 = row + kbase + 16 + 8 * hf;
#pragma unroll
  for (int i = 0; i < 8; ++i) { a[i] = p0[i]; a[8 + i] = p1[i]; }
  return a;
}

__device__ __forceinline__ v16h load_b_f16(const _Float16* __restrict__ col, int kbase, int hf) {
  v16h b;
  const _Float16* p0 = col + kbase + 8 * hf;
  const _Float16* p1 = col + kbase + 16 + 8 * hf;
#pragma unroll
  for (int j = 0; j < 8; ++j) { b[j] = p0[j]; b[8 + j] = p1[j]; }
  return b;
}

// ---------------------------------------------------------------------------
// One-time fp32 -> f16 weight conversion (W is tiny: 128*(512+768+768) elems)
// ---------------------------------------------------------------------------
__global__ __launch_bounds__(256) void cvt_f16_kernel(
    const float* __restrict__ src, _Float16* __restrict__ dst, int n)
{
  int i = blockIdx.x * blockDim.x + threadIdx.x;
  if (i < n) dst[i] = (_Float16)src[i];
}

// ---------------------------------------------------------------------------
// Projection GEMM: Y[m,n] = sum_i X[m,i] * Wh[n,i], n in [0,128).
// One wave -> 16 rows x 128 cols (8 accumulators). X fp32, Wh f16, Y f16.
// Software-pipelined: k-step (i+1) fragment loads issued before k-step (i)
// WMMAs so global loads stay in flight across the whole WMMA batch.
// transposeOut: store Y as [B, 128, L] (for V^T) instead of [M, 128].
// ---------------------------------------------------------------------------
__global__ __launch_bounds__(256) void proj_gemm_kernel(
    const float* __restrict__ X, const _Float16* __restrict__ Wh,
    _Float16* __restrict__ Y, int IC, int transposeOut, int L)
{
  const int lane = threadIdx.x & 31;
  const int wave = threadIdx.x >> 5;
  const int hf   = lane >> 4;
  const int c16  = lane & 15;
  const int row0 = (blockIdx.x * 8 + wave) * 16;

  v8f acc[8];
  const v8f vz = {0.f,0.f,0.f,0.f,0.f,0.f,0.f,0.f};
#pragma unroll
  for (int t = 0; t < 8; ++t) acc[t] = vz;

  const float* xrow = X + (size_t)(row0 + c16) * IC;
  const _Float16* wrow[8];
#pragma unroll
  for (int t = 0; t < 8; ++t) wrow[t] = Wh + (size_t)(16 * t + c16) * IC;

  // prologue: fragments for k = 0
  v16h a = load_a_f32(xrow, 0, hf);
  v16h bw[8];
#pragma unroll
  for (int t = 0; t < 8; ++t) bw[t] = load_b_f16(wrow[t], 0, hf);

  for (int k = 0; k < IC; k += 32) {
    // issue next k-step's loads (branchless wrap on last iter)
    const int kn = (k + 32 < IC) ? (k + 32) : 0;
    v16h a_n = load_a_f32(xrow, kn, hf);
    v16h bw_n[8];
#pragma unroll
    for (int t = 0; t < 8; ++t) bw_n[t] = load_b_f16(wrow[t], kn, hf);

    // compute with current fragments while next loads are in flight
#pragma unroll
    for (int t = 0; t < 8; ++t) acc[t] = wmma_f16(a, bw[t], acc[t]);

    a = a_n;
#pragma unroll
    for (int t = 0; t < 8; ++t) bw[t] = bw_n[t];
  }

  // C/D layout: element r of lane -> M = r + 8*hf, N = lane%16
#pragma unroll
  for (int t = 0; t < 8; ++t) {
#pragma unroll
    for (int r = 0; r < 8; ++r) {
      int m = row0 + r + 8 * hf;
      int n = 16 * t + c16;
      _Float16 v = (_Float16)acc[t][r];
      if (!transposeOut) {
        Y[(size_t)m * D_MODEL + n] = v;
      } else {
        int bb = m / L, pos = m % L;
        Y[((size_t)bb * D_MODEL + n) * L + pos] = v;
      }
    }
  }
}

// ---------------------------------------------------------------------------
// Fused flash attention: one wave owns a 16-row Q tile, streams KV in
// 32-wide tiles with online softmax. Q/K: [B,L,128] f16; Vt: [B,128,Lkv] f16.
// Pipelining: next tile's K fragments + current tile's Vt fragments are all
// issued before the softmax, so they overlap ~300 VALU ops of exp/shuffles.
// ---------------------------------------------------------------------------
__global__ __launch_bounds__(256) void attn_kernel(
    const _Float16* __restrict__ Qh, const _Float16* __restrict__ Kh,
    const _Float16* __restrict__ Vt, float* __restrict__ out,
    int B, int Lq, int Lkv)
{
  __shared__ _Float16 ptile[8][16 * 32];   // per-wave P staging (1 KB each)

  const int lane = threadIdx.x & 31;
  const int wave = threadIdx.x >> 5;
  const int hf   = lane >> 4;
  const int c16  = lane & 15;

  const int qtiles = Lq >> 4;
  const int gw = blockIdx.x * 8 + wave;
  const int b  = gw / qtiles;
  const int q0 = (gw % qtiles) * 16;

  const float scale = 0.088388347648318447f;   // 1/sqrt(128)

  // Q fragments for the whole d=128 (4 chunks of K=32), loaded once.
  v16h aq[4];
  const _Float16* qrow = Qh + ((size_t)b * Lq + q0 + c16) * D_MODEL;
#pragma unroll
  for (int kk = 0; kk < 4; ++kk) aq[kk] = load_a_f16(qrow, kk * 32, hf);

  const v8f vz = {0.f,0.f,0.f,0.f,0.f,0.f,0.f,0.f};
  v8f o[8];
#pragma unroll
  for (int t = 0; t < 8; ++t) o[t] = vz;
  float mrow[8], lrow[8];
#pragma unroll
  for (int r = 0; r < 8; ++r) { mrow[r] = -1e30f; lrow[r] = 0.f; }

  _Float16* lp = &ptile[wave][0];
  const _Float16* kbb = Kh + ((size_t)b * Lkv + c16) * D_MODEL;
  const _Float16* vbb = Vt + ((size_t)b * D_MODEL + c16) * Lkv;

  // prologue: K fragments for kv0 = 0 (frag 2*kk+tile, kk = d-chunk)
  v16h kf[8];
#pragma unroll
  for (int kk = 0; kk < 4; ++kk) {
    kf[2 * kk]     = load_b_f16(kbb,                kk * 32, hf);
    kf[2 * kk + 1] = load_b_f16(kbb + 16 * D_MODEL, kk * 32, hf);
  }

  for (int kv0 = 0; kv0 < Lkv; kv0 += 32) {
    // ---- S = Q * K^T for two 16-col tiles (fragments already resident) ---
    v8f s0 = vz, s1 = vz;
#pragma unroll
    for (int kk = 0; kk < 4; ++kk) {
      s0 = wmma_f16(aq[kk], kf[2 * kk],     s0);
      s1 = wmma_f16(aq[kk], kf[2 * kk + 1], s1);
    }

    // ---- issue next K tile + current Vt tile loads (overlap softmax) ----
    const int kvn = (kv0 + 32 < Lkv) ? (kv0 + 32) : 0;
    const _Float16* kb0 = kbb + (size_t)kvn * D_MODEL;
    v16h kf_n[8];
#pragma unroll
    for (int kk = 0; kk < 4; ++kk) {
      kf_n[2 * kk]     = load_b_f16(kb0,                kk * 32, hf);
      kf_n[2 * kk + 1] = load_b_f16(kb0 + 16 * D_MODEL, kk * 32, hf);
    }
    v16h vf[8];
#pragma unroll
    for (int t = 0; t < 8; ++t)
      vf[t] = load_b_f16(vbb + (size_t)(16 * t) * Lkv + kv0, 0, hf);

    // ---- online softmax (rows live across the 16-lane group) ------------
    float alpha[8];
#pragma unroll
    for (int r = 0; r < 8; ++r) {
      float s0r = s0[r] * scale;
      float s1r = s1[r] * scale;
      float mr = fmaxf(s0r, s1r);
#pragma unroll
      for (int off = 1; off < 16; off <<= 1)
        mr = fmaxf(mr, __shfl_xor(mr, off, 32));
      float mnew = fmaxf(mrow[r], mr);
      float av = __expf(mrow[r] - mnew);
      float p0 = __expf(s0r - mnew);
      float p1 = __expf(s1r - mnew);
      float rs = p0 + p1;
#pragma unroll
      for (int off = 1; off < 16; off <<= 1)
        rs += __shfl_xor(rs, off, 32);
      lrow[r] = lrow[r] * av + rs;
      mrow[r] = mnew;
      alpha[r] = av;
      // stage P (C-layout -> LDS [16][32]); per-wave region, DS in-order
      int m = r + 8 * hf;
      lp[m * 32 + c16]      = (_Float16)p0;
      lp[m * 32 + 16 + c16] = (_Float16)p1;
    }

    // rescale running O by alpha (row r+8*hf lives in element r)
#pragma unroll
    for (int t = 0; t < 8; ++t)
#pragma unroll
      for (int r = 0; r < 8; ++r) o[t][r] *= alpha[r];

    // ---- P as A-fragment from LDS ---------------------------------------
    v16h ap;
    const _Float16* prow = lp + c16 * 32;
#pragma unroll
    for (int i = 0; i < 8; ++i) {
      ap[i]     = prow[8 * hf + i];
      ap[8 + i] = prow[16 + 8 * hf + i];
    }

    // ---- O += P * V (Vt fragments already in flight / resident) ---------
#pragma unroll
    for (int t = 0; t < 8; ++t) o[t] = wmma_f16(ap, vf[t], o[t]);

    // rotate K double buffer
#pragma unroll
    for (int i = 0; i < 8; ++i) kf[i] = kf_n[i];
  }

  // ---- epilogue: out = O / l ---------------------------------------------
#pragma unroll
  for (int r = 0; r < 8; ++r) {
    float inv = 1.0f / lrow[r];
    int m = q0 + r + 8 * hf;
    float* orow = out + ((size_t)b * Lq + m) * D_MODEL;
#pragma unroll
    for (int t = 0; t < 8; ++t) orow[16 * t + c16] = o[t][r] * inv;
  }
}

// ---------------------------------------------------------------------------
// Launch: convert weights to f16, 3 projection GEMMs into f16 workspace
// (V transposed), then fused attention.
// Workspace: Q/K/Vt f16 = 12.6 MB + Wh f16 = 0.5 MB.
// ---------------------------------------------------------------------------
extern "C" void kernel_launch(void* const* d_in, const int* in_sizes, int n_in,
                              void* d_out, int out_size, void* d_ws, size_t ws_size,
                              hipStream_t stream) {
  (void)in_sizes; (void)n_in; (void)out_size; (void)ws_size;
  const float* q_input = (const float*)d_in[0];
  const float* k_input = (const float*)d_in[1];
  const float* v_input = (const float*)d_in[2];
  const float* Wq      = (const float*)d_in[3];
  const float* Wk      = (const float*)d_in[4];
  const float* Wv      = (const float*)d_in[5];
  float* out = (float*)d_out;

  const int B = 8, Lq = 2048, Lkv = 2048, QD = 512, KD = 768;

  _Float16* Qh  = (_Float16*)d_ws;
  _Float16* Kh  = Qh  + (size_t)B * Lq  * D_MODEL;
  _Float16* Vt  = Kh  + (size_t)B * Lkv * D_MODEL;
  _Float16* Wqh = Vt  + (size_t)B * Lkv * D_MODEL;
  _Float16* Wkh = Wqh + (size_t)D_MODEL * QD;
  _Float16* Wvh = Wkh + (size_t)D_MODEL * KD;

  const int nWq = D_MODEL * QD, nWk = D_MODEL * KD;
  cvt_f16_kernel<<<(nWq + 255) / 256, 256, 0, stream>>>(Wq, Wqh, nWq);
  cvt_f16_kernel<<<(nWk + 255) / 256, 256, 0, stream>>>(Wk, Wkh, nWk);
  cvt_f16_kernel<<<(nWk + 255) / 256, 256, 0, stream>>>(Wv, Wvh, nWk);

  const int Mq = B * Lq;   // 16384 rows; 128 rows per block (8 waves x 16)
  dim3 blk(256);
  proj_gemm_kernel<<<Mq / 128, blk, 0, stream>>>(q_input, Wqh, Qh, QD, 0, Lq);
  proj_gemm_kernel<<<Mq / 128, blk, 0, stream>>>(k_input, Wkh, Kh, KD, 0, Lkv);
  proj_gemm_kernel<<<Mq / 128, blk, 0, stream>>>(v_input, Wvh, Vt, KD, 1, Lkv);

  attn_kernel<<<(B * (Lq / 16)) / 8, blk, 0, stream>>>(Qh, Kh, Vt, out, B, Lq, Lkv);
}
